// TransformerAggregator_20237885898832
// MI455X (gfx1250) — matrix operand order repaired and checked by reading, not compiled, not run on hardware
//
#include <hip/hip_runtime.h>
#include <hip/hip_bf16.h>

// ---------------------------------------------------------------------------
// TransformerAggregator for MI455X (gfx1250), bf16 WMMA path.
//   D=512, A=V=FFH=256, H=1, B=4096, S=50  -> M = B*S = 204800 tokens.
// All GEMMs: bf16 A/B, f32 accumulate via v_wmma_f32_16x16x32_bf16.
// A fragments load as two aligned 16B vectors; B fragments are gathered per element.
// LayerNorm / softmax / pooling in fp32.
// ---------------------------------------------------------------------------

typedef __bf16 bf16;
typedef __bf16 v16bf __attribute__((ext_vector_type(16)));
typedef __bf16 v8bf  __attribute__((ext_vector_type(8)));
typedef __bf16 v2bf  __attribute__((ext_vector_type(2)));
typedef float  v8f   __attribute__((ext_vector_type(8)));

#define WMMA_BF16(a, b, c) \
  __builtin_amdgcn_wmma_f32_16x16x32_bf16(false, (a), false, (b), (short)0, (c), false, false)

static constexpr int Bn   = 4096;
static constexpr int Sn   = 50;
static constexpr int Dn   = 512;
static constexpr int An   = 256;
static constexpr int Mtok = Bn * Sn;      // 204800
static constexpr int SP   = 64;           // padded sequence length

// --- A fragment: 16x32 bf16 tile, row-major, leading dim lda (lda even,
// tile base 32B-aligned). ISA 7.12.2: lane m holds k{0..7} and k{16..23}
// (lane-half adds +8) -> two contiguous 16-byte runs.
__device__ inline v16bf load_a16x32(const bf16* __restrict__ A, int lda) {
  const int lane = threadIdx.x & 31;
  const int m    = lane & 15;
  const int hh   = lane >> 4;
  const bf16* r  = A + (size_t)m * lda + 8 * hh;
  v8bf lo = *(const v8bf*)(r);        // k 0..7   (+8*hh)
  v8bf hi = *(const v8bf*)(r + 16);   // k 16..23 (+8*hh)
  return __builtin_shufflevector(lo, hi, 0, 1, 2, 3, 4, 5, 6, 7,
                                         8, 9, 10, 11, 12, 13, 14, 15);
}

// --- B fragment: 32x16 bf16 tile, row-major K x N, leading dim ldb.
// lane holds column n = lane%16; elements k{0..7},k{16..23} (lane-half adds +8),
// matching the A fragment k order (strided scalar gather).
__device__ inline v16bf load_b32x16(const bf16* __restrict__ Bm, int ldb) {
  const int lane = threadIdx.x & 31;
  const int n    = lane & 15;
  const int hh   = lane >> 4;
  v16bf v;
#pragma unroll
  for (int s = 0; s < 8; ++s) {
    v[s]     = Bm[(size_t)(8 * hh + s) * ldb + n];
    v[8 + s] = Bm[(size_t)(16 + 8 * hh + s) * ldb + n];
  }
  return v;
}

// ---------------------------------------------------------------------------
// fp32 -> bf16 conversion (weights)
__global__ __launch_bounds__(256) void cvt_kernel(const float* __restrict__ s,
                                                  bf16* __restrict__ d, int n) {
  int i = blockIdx.x * 256 + threadIdx.x;
  if (i < n) d[i] = (bf16)s[i];
}

// Zero the 16-row pad after the qkv activation buffer (rows Mtok..Mtok+15).
__global__ __launch_bounds__(256) void zero_pad_kernel(bf16* __restrict__ q) {
  int i = blockIdx.x * 256 + threadIdx.x;
  if (i < 16 * 768) q[(size_t)Mtok * 768 + i] = (bf16)0.0f;
}

// Embedding gather -> bf16 x  (one block per token, paired stores)
__global__ __launch_bounds__(256) void gather_kernel(const float* __restrict__ T,
                                                     const int* __restrict__ idx,
                                                     bf16* __restrict__ X) {
  const size_t t   = blockIdx.x;
  const float* src = T + (size_t)idx[t] * Dn;
  bf16* dst        = X + t * Dn;
  const int d = threadIdx.x * 2;   // 256 threads x 2 = 512
  v2bf o;
  o[0] = (bf16)src[d];
  o[1] = (bf16)src[d + 1];
  *(v2bf*)(dst + d) = o;
}

// ---------------------------------------------------------------------------
// Generic GEMM: C[M,ND] = act(A[M,KD] @ W[KD,ND] + bias). TILE_M = 128.
// 8 waves/block, wave w owns rows [w*16, w*16+16); weights stream from L2.
// WRITE_KT: additionally scatter columns [256,512) transposed per batch into
// kT[b][dim][SP] (token-padded to 64) for the attention scores B operand.
template <int KD, int ND, bool RELU, bool WRITE_KT>
__global__ __launch_bounds__(256) void gemm_bias_kernel(const bf16* __restrict__ A,
                                                        const bf16* __restrict__ W,
                                                        const float* __restrict__ bias,
                                                        bf16* __restrict__ C,
                                                        bf16* __restrict__ kT) {
  const int wv    = threadIdx.x >> 5;
  const int lane  = threadIdx.x & 31;
  const size_t m0 = (size_t)blockIdx.x * 128 + wv * 16;
  const int nlo   = lane & 15;
  const int hh    = lane >> 4;
#pragma unroll 1
  for (int nt = 0; nt < ND / 16; ++nt) {
    v8f acc = {0, 0, 0, 0, 0, 0, 0, 0};
#pragma unroll 4
    for (int k0 = 0; k0 < KD; k0 += 32) {
      v16bf a = load_a16x32(A + m0 * KD + k0, KD);
      v16bf b = load_b32x16(W + (size_t)k0 * ND + nt * 16, ND);
      acc = WMMA_BF16(a, b, acc);
    }
    const int n    = nt * 16 + nlo;
    const float bv = bias[n];
    const bool kcol = WRITE_KT && (nt >= 16) && (nt < 32);  // uniform per nt
#pragma unroll
    for (int j = 0; j < 8; ++j) {
      float v = acc[j] + bv;
      if (RELU) v = fmaxf(v, 0.0f);
      const size_t row = m0 + j + 8 * hh;
      C[row * ND + n] = (bf16)v;
      if (kcol) {
        const int t  = (int)row;
        const int bb = t / Sn;
        const int s  = t - bb * Sn;
        kT[((size_t)bb * An + (n - An)) * SP + s] = (bf16)v;
      }
    }
  }
}

// ---------------------------------------------------------------------------
// GEMM + bias + residual + LayerNorm, ND fixed to 512 (full rows in-block).
// TILE_M = 32 rows; fp32 row tile staged in LDS (64KB), 8-thread row teams.
template <int KD>
__global__ __launch_bounds__(256) void gemm_res_ln_kernel(const bf16* __restrict__ A,
                                                          const bf16* __restrict__ W,
                                                          const float* __restrict__ bias,
                                                          const bf16* __restrict__ R,
                                                          const float* __restrict__ gamma,
                                                          const float* __restrict__ beta,
                                                          bf16* __restrict__ Out) {
  constexpr int ND = 512;
  __shared__ float sT[32][ND];
  __shared__ float red[32][8];
  const int tid   = threadIdx.x;
  const int wv    = tid >> 5;
  const int lane  = tid & 31;
  const size_t m0 = (size_t)blockIdx.x * 32;
  const int mt    = wv >> 2;      // 0..1 : which 16-row half
  const int ntB   = wv & 3;       // n-tile stride 4
  const int nlo   = lane & 15;
  const int hh    = lane >> 4;
  const size_t mrow = m0 + mt * 16;

  for (int nt = ntB; nt < ND / 16; nt += 4) {
    v8f acc = {0, 0, 0, 0, 0, 0, 0, 0};
#pragma unroll 4
    for (int k0 = 0; k0 < KD; k0 += 32) {
      v16bf a = load_a16x32(A + mrow * KD + k0, KD);
      v16bf b = load_b32x16(W + (size_t)k0 * ND + nt * 16, ND);
      acc = WMMA_BF16(a, b, acc);
    }
    const int n    = nt * 16 + nlo;
    const float bv = bias[n];
#pragma unroll
    for (int j = 0; j < 8; ++j) {
      const int lr = mt * 16 + j + 8 * hh;
      sT[lr][n] = acc[j] + bv + (float)R[(m0 + lr) * ND + n];
    }
  }
  __syncthreads();

  // LayerNorm: 32 rows x 8 partials, 64 cols each.
  const int row = tid >> 3, part = tid & 7;
  float sum = 0.0f;
  for (int c = part * 64; c < part * 64 + 64; ++c) sum += sT[row][c];
  red[row][part] = sum;
  __syncthreads();
  float mu = 0.0f;
#pragma unroll
  for (int p = 0; p < 8; ++p) mu += red[row][p];
  mu *= (1.0f / ND);
  __syncthreads();
  float vs = 0.0f;
  for (int c = part * 64; c < part * 64 + 64; ++c) {
    const float d = sT[row][c] - mu;
    vs += d * d;
  }
  red[row][part] = vs;
  __syncthreads();
  float var = 0.0f;
#pragma unroll
  for (int p = 0; p < 8; ++p) var += red[row][p];
  const float rstd = rsqrtf(var * (1.0f / ND) + 1e-6f);
  for (int c = part * 64; c < part * 64 + 64; ++c) {
    const float o = (sT[row][c] - mu) * rstd * gamma[c] + beta[c];
    Out[(m0 + row) * ND + c] = (bf16)o;
  }
}

// ---------------------------------------------------------------------------
// Attention (H=1): one block per sequence; S=50 padded to 64.
// q (A-frags) and v (B-frags) come straight from the qkv buffer; k comes from
// the pre-transposed kT buffer. Padded rows read the zeroed qkv pad; masked
// columns (-1e9 -> softmax 0) make padded keys inert.
__global__ __launch_bounds__(256) void attention_kernel(const bf16* __restrict__ qkv,
                                                        const bf16* __restrict__ kT,
                                                        const int* __restrict__ lengths,
                                                        bf16* __restrict__ ctx) {
  __shared__ float sc[SP][SP];
  __shared__ bf16 at[SP][SP];
  __shared__ float pr[SP][4];

  const int b   = blockIdx.x;
  const int tid = threadIdx.x;
  const int len = lengths[b];
  const float qscale = 0.04419417382415922f;  // 1/sqrt(D/H) = 1/sqrt(512)

  const int wv   = tid >> 5;
  const int lane = tid & 31;
  const int nlo  = lane & 15;
  const int hh   = lane >> 4;

  const bf16* qbase = qkv + (size_t)b * Sn * 768;        // q columns [0,256)
  const bf16* vbase = qbase + 2 * An;                    // v columns [512,768)
  const bf16* ktb   = kT + (size_t)b * An * SP;

  // scores = q @ k^T * qscale : 4x4 16x16 tiles, 2 per wave
  {
    const int mt = wv >> 1;
#pragma unroll
    for (int t = 0; t < 2; ++t) {
      const int nt = (wv & 1) * 2 + t;
      v8f acc = {0, 0, 0, 0, 0, 0, 0, 0};
#pragma unroll
      for (int k0 = 0; k0 < An; k0 += 32) {
        v16bf a   = load_a16x32(qbase + (size_t)(mt * 16) * 768 + k0, 768);
        v16bf bfr = load_b32x16(ktb + (size_t)k0 * SP + nt * 16, SP);
        acc = WMMA_BF16(a, bfr, acc);
      }
      const int n = nt * 16 + nlo;
      const bool valid = (n <= len) && (n < Sn);
#pragma unroll
      for (int j = 0; j < 8; ++j)
        sc[mt * 16 + j + 8 * hh][n] = valid ? acc[j] * qscale : -1.0e9f;
    }
  }
  __syncthreads();

  // row softmax: 64 rows x 4-thread teams
  {
    const int row = tid >> 2, part = tid & 3;
    float mx = -3.0e38f;
    for (int c = part * 16; c < part * 16 + 16; ++c) mx = fmaxf(mx, sc[row][c]);
    pr[row][part] = mx;
    __syncthreads();
    mx = fmaxf(fmaxf(pr[row][0], pr[row][1]), fmaxf(pr[row][2], pr[row][3]));
    __syncthreads();
    float sum = 0.0f;
    for (int c = part * 16; c < part * 16 + 16; ++c) {
      const float e = __expf(sc[row][c] - mx);
      sc[row][c] = e;
      sum += e;
    }
    pr[row][part] = sum;
    __syncthreads();
    sum = pr[row][0] + pr[row][1] + pr[row][2] + pr[row][3];
    const float inv = 1.0f / sum;
    for (int c = part * 16; c < part * 16 + 16; ++c)
      at[row][c] = (bf16)(sc[row][c] * inv);
  }
  __syncthreads();

  // ctx = attn @ v : 4x16 tiles, 8 per wave
#pragma unroll
  for (int t = 0; t < 2; ++t) {
    const int nt = wv + t * 8;
#pragma unroll 1
    for (int mt = 0; mt < 4; ++mt) {
      v8f acc = {0, 0, 0, 0, 0, 0, 0, 0};
#pragma unroll
      for (int k0 = 0; k0 < SP; k0 += 32) {
        v16bf a   = load_a16x32(&at[mt * 16][k0], SP);
        v16bf bfr = load_b32x16(vbase + (size_t)k0 * 768 + nt * 16, 768);
        acc = WMMA_BF16(a, bfr, acc);
      }
      const int n = nt * 16 + nlo;
#pragma unroll
      for (int j = 0; j < 8; ++j) {
        const int srow = mt * 16 + j + 8 * hh;
        if (srow < Sn) ctx[((size_t)b * Sn + srow) * An + n] = (bf16)acc[j];
      }
    }
  }
}

// Masked mean-pool: grid (B, 2), 256 threads -> 512 dims
__global__ __launch_bounds__(256) void pool_kernel(const bf16* __restrict__ X,
                                                   const int* __restrict__ lengths,
                                                   float* __restrict__ P) {
  const int b = blockIdx.x;
  const int d = blockIdx.y * 256 + threadIdx.x;
  const int len = lengths[b];
  float s = 0.0f;
  for (int t = 0; t <= len; ++t) s += (float)X[((size_t)b * Sn + t) * Dn + d];
  P[(size_t)b * Dn + d] = s / (float)(len + 1);
}

// ---------------------------------------------------------------------------
extern "C" void kernel_launch(void* const* d_in, const int* in_sizes, int n_in,
                              void* d_out, int out_size, void* d_ws, size_t ws_size,
                              hipStream_t stream) {
  const float* embed = (const float*)d_in[0];
  const int*   idx   = (const int*)d_in[1];
  const int*   lens  = (const int*)d_in[2];
  const float* W1    = (const float*)d_in[3];
  const float* b1    = (const float*)d_in[4];
  const float* W2    = (const float*)d_in[5];
  const float* b2    = (const float*)d_in[6];
  const float* ln1g  = (const float*)d_in[7];
  const float* ln1b  = (const float*)d_in[8];
  const float* Wqkv  = (const float*)d_in[9];
  const float* bqkv  = (const float*)d_in[10];
  const float* Wo    = (const float*)d_in[11];
  const float* bo    = (const float*)d_in[12];
  const float* ln2g  = (const float*)d_in[13];
  const float* ln2b  = (const float*)d_in[14];
  float* out = (float*)d_out;

  char* ws = (char*)d_ws;
  size_t off = 0;
  auto carve = [&](size_t bytes) {
    char* p = ws + off;
    off += (bytes + 255) & ~(size_t)255;
    return p;
  };
  bf16* w1b   = (bf16*)carve((size_t)Dn * 256 * 2);
  bf16* w2b   = (bf16*)carve((size_t)256 * Dn * 2);
  bf16* wqkvb = (bf16*)carve((size_t)Dn * 768 * 2);
  bf16* wob   = (bf16*)carve((size_t)256 * Dn * 2);
  bf16* x_bf  = (bf16*)carve((size_t)Mtok * Dn * 2);
  bf16* h1_bf = (bf16*)carve((size_t)Mtok * 256 * 2);
  bf16* ff_bf = (bf16*)carve((size_t)Mtok * Dn * 2);
  bf16* qk_bf = (bf16*)carve((size_t)(Mtok + 16) * 768 * 2);   // +16 pad rows
  bf16* kT_bf = (bf16*)carve((size_t)Bn * An * SP * 2);        // k transposed
  bf16* cx_bf = (bf16*)carve((size_t)Mtok * 256 * 2);
  bf16* ot_bf = (bf16*)carve((size_t)Mtok * Dn * 2);
  (void)ws_size; (void)n_in; (void)in_sizes; (void)out_size;

  // Weights -> bf16 (L2-resident thereafter); zero qkv pad rows.
  cvt_kernel<<<(Dn * 256 + 255) / 256, 256, 0, stream>>>(W1, w1b, Dn * 256);
  cvt_kernel<<<(256 * Dn + 255) / 256, 256, 0, stream>>>(W2, w2b, 256 * Dn);
  cvt_kernel<<<(Dn * 768 + 255) / 256, 256, 0, stream>>>(Wqkv, wqkvb, Dn * 768);
  cvt_kernel<<<(256 * Dn + 255) / 256, 256, 0, stream>>>(Wo, wob, 256 * Dn);
  zero_pad_kernel<<<(16 * 768 + 255) / 256, 256, 0, stream>>>(qk_bf);

  // Embedding gather
  gather_kernel<<<Mtok, 256, 0, stream>>>(embed, idx, x_bf);

  // h1 = relu(x@W1 + b1)
  gemm_bias_kernel<Dn, 256, true, false>
      <<<Mtok / 128, 256, 0, stream>>>(x_bf, w1b, b1, h1_bf, nullptr);
  // ff = LN1(h1@W2 + b2 + x)
  gemm_res_ln_kernel<256>
      <<<Mtok / 32, 256, 0, stream>>>(h1_bf, w2b, b2, x_bf, ln1g, ln1b, ff_bf);
  // qkv = ff@Wqkv + bqkv   (also emits kT)
  gemm_bias_kernel<Dn, 768, false, true>
      <<<Mtok / 128, 256, 0, stream>>>(ff_bf, wqkvb, bqkv, qk_bf, kT_bf);
  // attention -> ctx
  attention_kernel<<<Bn, 256, 0, stream>>>(qk_bf, kT_bf, lens, cx_bf);
  // out = LN2(ctx@Wo + bo + ff)
  gemm_res_ln_kernel<256>
      <<<Mtok / 32, 256, 0, stream>>>(cx_bf, wob, bo, ff_bf, ln2g, ln2b, ot_bf);
  // masked mean pool
  pool_kernel<<<dim3(Bn, 2), 256, 0, stream>>>(ot_bf, lens, out);
}
